// SimpleGAT_55362128445712
// MI455X (gfx1250) — compile-verified, matrix operand order checked by reading, not run on hardware
//
#include <hip/hip_runtime.h>
#include <math.h>

// ---------------- problem constants (match reference) ----------------
#define NNODES 50000
#define FIN    128
#define HIDC   32
#define NHEAD  4
#define NCLSK  10
#define NGRAPH 64
#define SLOPE  0.2f

typedef __attribute__((ext_vector_type(16))) _Float16 v16h;
typedef __attribute__((ext_vector_type(8)))  _Float16 v8h;
typedef __attribute__((ext_vector_type(8)))  float    v8f;
typedef __attribute__((ext_vector_type(4)))  float    v4f;

// ---------------------------------------------------------------------
// Weight prep: BT16[n*K + k] = (f16) B[k*Nc + n]   (transpose + convert)
// Weights are <=128KB, stay L2/L0 resident across the whole GEMM.
// ---------------------------------------------------------------------
__global__ void convert_bt_kernel(const float* __restrict__ B, _Float16* __restrict__ BT,
                                  int K, int Nc)
{
  int t = blockIdx.x * blockDim.x + threadIdx.x;
  if (t >= K * Nc) return;
  int n = t / K, k = t % K;
  BT[t] = (_Float16)B[(size_t)k * Nc + n];
}

// ---------------------------------------------------------------------
// WMMA GEMM: C[M,Nc] = A[M,K](f32) * B[K,Nc] (+bias), B given as BT16[Nc,K] f16.
// One wave32 computes a 16x64 strip: one A fragment feeds 4 WMMAs.
// Requires M % 16 == 0 and Nc % 64 == 0 (true for all calls here).
//
// ISA 7.12.2 16-bit A 16x32 layout: lane l owns row m=l&15; its 16 halves are
// K = khalf*8 + [0..7] and 16 + khalf*8 + [0..7]  (khalf = l>>4)  -> two
// contiguous 8-element runs, so A loads are 4x b128 f32 and B loads are
// 2x b128 f16 from the transposed weight buffer. No divergent guards.
// ---------------------------------------------------------------------
__global__ __launch_bounds__(256)
void wmma_gemm_kernel(const float* __restrict__ A, const _Float16* __restrict__ BT,
                      const float* __restrict__ bias, float* __restrict__ C,
                      int M, int K, int Nc)
{
  const int lane    = threadIdx.x & 31;
  const int wave    = threadIdx.x >> 5;
  const int stripsN = Nc >> 6;                               // 64-wide strips
  const int waveId  = blockIdx.x * (blockDim.x >> 5) + wave; // uniform per wave
  if (waveId >= (M >> 4) * stripsN) return;                  // wave-uniform exit
  const int tileM   = waveId / stripsN;
  const int colBase = (waveId % stripsN) << 6;

  const int l15   = lane & 15;
  const int khalf = lane >> 4;
  const int mrow  = (tileM << 4) + l15;

  const float*    arow  = A  + (size_t)mrow * K + (khalf << 3);
  const _Float16* bbase = BT + (size_t)(colBase + l15) * K + (khalf << 3);

  v8f acc[4] = {{}, {}, {}, {}};

  for (int k0 = 0; k0 < K; k0 += 32) {
    // ---- A fragment: 4x b128 loads + f32->f16 convert ----
    v4f a0 = *(const v4f*)(arow + k0);
    v4f a1 = *(const v4f*)(arow + k0 + 4);
    v4f a2 = *(const v4f*)(arow + k0 + 16);
    v4f a3 = *(const v4f*)(arow + k0 + 20);
    v16h fa;
#pragma unroll
    for (int i = 0; i < 4; ++i) {
      fa[i]      = (_Float16)a0[i];
      fa[4 + i]  = (_Float16)a1[i];
      fa[8 + i]  = (_Float16)a2[i];
      fa[12 + i] = (_Float16)a3[i];
    }
    // ---- 4 column tiles reuse the same A fragment ----
#pragma unroll
    for (int j = 0; j < 4; ++j) {
      const _Float16* bp = bbase + ((size_t)(j << 4)) * K + k0;
      v8h b0 = *(const v8h*)(bp);
      v8h b1 = *(const v8h*)(bp + 16);
      v16h fb = __builtin_shufflevector(b0, b1, 0, 1, 2, 3, 4, 5, 6, 7,
                                        8, 9, 10, 11, 12, 13, 14, 15);
      acc[j] = __builtin_amdgcn_wmma_f32_16x16x32_f16(
          false, fa, false, fb, (short)0, acc[j], false, false);
    }
  }

  // ---- epilogue: bias + store ----
#pragma unroll
  for (int j = 0; j < 4; ++j) {
    const int n  = colBase + (j << 4) + l15;
    const float bv = bias ? bias[n] : 0.f;
#pragma unroll
    for (int r = 0; r < 8; ++r) {
      const int m = (tileM << 4) + r + (khalf << 3);
      C[(size_t)m * Nc + n] = acc[j][r] + bv;
    }
  }
}

// ---------------- scalar / atomic helper kernels ----------------
__global__ void fill_f32_kernel(float* __restrict__ p, float v, int n) {
  int i = blockIdx.x * blockDim.x + threadIdx.x;
  if (i < n) p[i] = v;
}

__device__ __forceinline__ float lrelu(float x) { return x > 0.f ? x : SLOPE * x; }

__device__ __forceinline__ void atomicMaxF(float* addr, float val) {
  if (val >= 0.f) atomicMax((int*)addr, __float_as_int(val));
  else            atomicMin((unsigned int*)addr, __float_as_uint(val));
}

// per-node attention scores: scs[n,h] = sum_c P[n,h*32+c]*a_s[h,c]; same for scd
__global__ void node_scores_kernel(const float* __restrict__ P,
                                   const float* __restrict__ a_s,
                                   const float* __restrict__ a_d,
                                   float* __restrict__ scs, float* __restrict__ scd)
{
  int t = blockIdx.x * blockDim.x + threadIdx.x;
  if (t >= NNODES * NHEAD) return;
  int n = t / NHEAD, h = t % NHEAD;
  const float* row = P + (size_t)n * (NHEAD * HIDC) + h * HIDC;
  float ss = 0.f, sd = 0.f;
#pragma unroll 8
  for (int c = 0; c < HIDC; ++c) {
    float x = row[c];
    ss += x * a_s[h * HIDC + c];
    sd += x * a_d[h * HIDC + c];
  }
  scs[t] = ss; scd[t] = sd;
}

__device__ __forceinline__ void edge_sd(const int* ei, int E, int e, int& s, int& d) {
  if (e < E) { s = ei[e]; d = ei[E + e]; }     // edge_index[0][e], edge_index[1][e]
  else       { s = d = e - E; }                // appended self loops
}

// segment max of leaky_relu(scs[src]+scd[dst]) into mbuf[dst,h]
__global__ void edge_max_kernel(const int* __restrict__ ei, int E, int Etot,
                                const float* __restrict__ scs,
                                const float* __restrict__ scd,
                                float* __restrict__ mbuf)
{
  int t = blockIdx.x * blockDim.x + threadIdx.x;
  if (t >= Etot * NHEAD) return;
  int e = t >> 2, h = t & 3, s, d;
  edge_sd(ei, E, e, s, d);
  atomicMaxF(&mbuf[d * NHEAD + h], lrelu(scs[s * NHEAD + h] + scd[d * NHEAD + h]));
}

// alpha numerator + denominator accumulation
__global__ void edge_exp_kernel(const int* __restrict__ ei, int E, int Etot,
                                const float* __restrict__ scs,
                                const float* __restrict__ scd,
                                const float* __restrict__ mbuf,
                                float* __restrict__ alp, float* __restrict__ den)
{
  int t = blockIdx.x * blockDim.x + threadIdx.x;
  if (t >= Etot * NHEAD) return;
  int e = t >> 2, h = t & 3, s, d;
  edge_sd(ei, E, e, s, d);
  float ee = lrelu(scs[s * NHEAD + h] + scd[d * NHEAD + h]);
  float al = expf(ee - mbuf[d * NHEAD + h]);
  alp[t] = al;
  atomicAdd(&den[d * NHEAD + h], al);
}

// scatter messages: agg[dst, h*32+c] += P[src, h*32+c] * alpha_norm
__global__ void edge_scatter_kernel(const int* __restrict__ ei, int E, int Etot,
                                    const float* __restrict__ P,
                                    const float* __restrict__ alp,
                                    const float* __restrict__ den,
                                    float* __restrict__ agg)
{
  long t = (long)blockIdx.x * blockDim.x + threadIdx.x;
  if (t >= (long)Etot * (NHEAD * HIDC)) return;
  int e = (int)(t >> 7), j = (int)(t & 127), h = j >> 5, s, d;
  edge_sd(ei, E, e, s, d);
  float al = alp[e * NHEAD + h] / fmaxf(den[d * NHEAD + h], 1e-16f);
  atomicAdd(&agg[(size_t)d * (NHEAD * HIDC) + j],
            P[(size_t)s * (NHEAD * HIDC) + j] * al);
}

// head mean + bias + ELU -> hn[N,32]
__global__ void finish_layer_kernel(const float* __restrict__ agg,
                                    const float* __restrict__ bias,
                                    float* __restrict__ hn)
{
  int t = blockIdx.x * blockDim.x + threadIdx.x;
  if (t >= NNODES * HIDC) return;
  int n = t >> 5, c = t & 31;
  const float* a = agg + (size_t)n * (NHEAD * HIDC);
  float v = (a[c] + a[HIDC + c] + a[2 * HIDC + c] + a[3 * HIDC + c]) * 0.25f + bias[c];
  hn[t] = v > 0.f ? v : (expf(v) - 1.f);
}

// global mean pool accumulation
__global__ void pool_accum_kernel(const float* __restrict__ hn,
                                  const int* __restrict__ batch,
                                  float* __restrict__ sums, float* __restrict__ cnt)
{
  int t = blockIdx.x * blockDim.x + threadIdx.x;
  if (t >= NNODES * HIDC) return;
  int n = t >> 5, c = t & 31;
  int g = batch[n];
  atomicAdd(&sums[g * HIDC + c], hn[t]);
  if (c == 0) atomicAdd(&cnt[g], 1.f);
}

// pooled @ lin1_w + lin1_b -> out[64,10]
__global__ void head_kernel(const float* __restrict__ sums, const float* __restrict__ cnt,
                            const float* __restrict__ w, const float* __restrict__ b,
                            float* __restrict__ out)
{
  int t = blockIdx.x * blockDim.x + threadIdx.x;
  if (t >= NGRAPH * NCLSK) return;
  int g = t / NCLSK, k = t % NCLSK;
  float cc = fmaxf(cnt[g], 1.f);
  float acc = b[k];
#pragma unroll
  for (int c = 0; c < HIDC; ++c) acc += (sums[g * HIDC + c] / cc) * w[c * NCLSK + k];
  out[t] = acc;
}

// ---------------------------------------------------------------------
extern "C" void kernel_launch(void* const* d_in, const int* in_sizes, int n_in,
                              void* d_out, int out_size, void* d_ws, size_t ws_size,
                              hipStream_t stream)
{
  const float* x      = (const float*)d_in[0];
  const int*   ei     = (const int*)  d_in[1];
  const int*   batch  = (const int*)  d_in[2];
  const float* enc1_w = (const float*)d_in[3];
  const float* enc1_b = (const float*)d_in[4];
  const float* enc2_w = (const float*)d_in[5];
  const float* enc2_b = (const float*)d_in[6];
  const float* lin1_w = (const float*)d_in[7];
  const float* lin1_b = (const float*)d_in[8];
  const float* gw [4] = {(const float*)d_in[ 9], (const float*)d_in[13], (const float*)d_in[17], (const float*)d_in[21]};
  const float* gas[4] = {(const float*)d_in[10], (const float*)d_in[14], (const float*)d_in[18], (const float*)d_in[22]};
  const float* gad[4] = {(const float*)d_in[11], (const float*)d_in[15], (const float*)d_in[19], (const float*)d_in[23]};
  const float* gb [4] = {(const float*)d_in[12], (const float*)d_in[16], (const float*)d_in[20], (const float*)d_in[24]};

  const int E    = in_sizes[1] / 2;
  const int Etot = E + NNODES;

  // ---- carve workspace (f32 units; every carve is a multiple of 4 floats,
  //      keeping 16B alignment for b128 loads) ----
  float* ws = (float*)d_ws;
  size_t off = 0;
  auto carve = [&](size_t n) { float* p = ws + off; off += n; return p; };
  float* h1   = carve((size_t)NNODES * 128);   // encoder 1 out
  float* h2   = carve((size_t)NNODES * 256);   // encoder 2 out
  float* P    = carve((size_t)NNODES * 128);   // per-layer projection [N,H,C]
  float* agg  = carve((size_t)NNODES * 128);   // message accumulator
  float* hn   = carve((size_t)NNODES * HIDC);  // layer output [N,32]
  float* scs  = carve((size_t)NNODES * NHEAD);
  float* scd  = carve((size_t)NNODES * NHEAD);
  float* mbuf = carve((size_t)NNODES * NHEAD);
  float* den  = carve((size_t)NNODES * NHEAD);
  float* alp  = carve((size_t)Etot * NHEAD);
  float* sums = carve((size_t)NGRAPH * HIDC);
  float* cnt  = carve((size_t)NGRAPH + 64);            // pad to keep alignment
  _Float16* bt = (_Float16*)carve((size_t)(256 * 128) / 2);  // transposed f16 weights
  (void)ws_size; (void)n_in; (void)out_size;

  auto blocks1d = [](long n, int bs) { return (unsigned)((n + bs - 1) / bs); };

  // convert/transpose weights, then WMMA GEMM (M%16==0, Nc%64==0 always here)
  auto gemm = [&](const float* A, const float* B, const float* bias, float* C,
                  int M, int K, int Nc) {
    convert_bt_kernel<<<blocks1d((long)K * Nc, 256), 256, 0, stream>>>(B, bt, K, Nc);
    long waves = (long)(M >> 4) * (Nc >> 6);
    wmma_gemm_kernel<<<blocks1d(waves * 32, 256), 256, 0, stream>>>(A, bt, bias, C, M, K, Nc);
  };

  // ---- encoders ----
  gemm(x,  enc1_w, enc1_b, h1, NNODES, 128, 128);
  gemm(h1, enc2_w, enc2_b, h2, NNODES, 128, 256);

  // ---- GAT layers ----
  auto gat = [&](const float* hin, int din, int li) {
    gemm(hin, gw[li], nullptr, P, NNODES, din, NHEAD * HIDC);
    fill_f32_kernel<<<blocks1d(NNODES * NHEAD, 256), 256, 0, stream>>>(mbuf, -INFINITY, NNODES * NHEAD);
    fill_f32_kernel<<<blocks1d(NNODES * NHEAD, 256), 256, 0, stream>>>(den, 0.f, NNODES * NHEAD);
    fill_f32_kernel<<<blocks1d((long)NNODES * 128, 256), 256, 0, stream>>>(agg, 0.f, NNODES * 128);
    node_scores_kernel<<<blocks1d(NNODES * NHEAD, 256), 256, 0, stream>>>(P, gas[li], gad[li], scs, scd);
    edge_max_kernel<<<blocks1d((long)Etot * NHEAD, 256), 256, 0, stream>>>(ei, E, Etot, scs, scd, mbuf);
    edge_exp_kernel<<<blocks1d((long)Etot * NHEAD, 256), 256, 0, stream>>>(ei, E, Etot, scs, scd, mbuf, alp, den);
    edge_scatter_kernel<<<blocks1d((long)Etot * 128, 256), 256, 0, stream>>>(ei, E, Etot, P, alp, den, agg);
    finish_layer_kernel<<<blocks1d((long)NNODES * HIDC, 256), 256, 0, stream>>>(agg, gb[li], hn);
  };

  gat(h2, 256, 0);
  gat(hn, HIDC, 1);
  gat(hn, HIDC, 2);
  gat(hn, HIDC, 3);

  // ---- pooling + classifier ----
  fill_f32_kernel<<<1, 256, 0, stream>>>(sums, 0.f, NGRAPH * HIDC);
  fill_f32_kernel<<<1, 64, 0, stream>>>(cnt, 0.f, NGRAPH);
  pool_accum_kernel<<<blocks1d((long)NNODES * HIDC, 256), 256, 0, stream>>>(hn, batch, sums, cnt);
  head_kernel<<<blocks1d(NGRAPH * NCLSK, 256), 256, 0, stream>>>(sums, cnt, lin1_w, lin1_b, (float*)d_out);
}
